// WaveFieldAttention_45835890983256
// MI455X (gfx1250) — compile-verified, matrix operand order checked by reading, not run on hardware
//
#include <hip/hip_runtime.h>

// ---------------------------------------------------------------- constants
constexpr int Bc  = 4;
constexpr int Nc  = 2048;
constexpr int Dc  = 1024;
constexpr int Hc  = 16;
constexpr int HDc = 64;
constexpr int Gc  = 2048;
constexpr int BN  = Bc * Nc;                       // 8192 rows
constexpr float STRIDEF = (float)(Gc - 1) / (float)(2048 - 1);  // MAX_SEQ = 2048

// ---------------------------------------------------------------- types
typedef __attribute__((ext_vector_type(16))) __bf16          v16bf;
typedef __attribute__((ext_vector_type(8)))  float           v8f;
typedef __attribute__((ext_vector_type(8)))  unsigned short  us8;
typedef __attribute__((ext_vector_type(4)))  unsigned short  us4;
typedef __attribute__((ext_vector_type(4)))  float           f4;
typedef __attribute__((ext_vector_type(4)))  unsigned int    u32x4;
typedef __attribute__((ext_vector_type(8)))  int             i32x8;
typedef __attribute__((ext_vector_type(4)))  int             i32x4;

#if defined(__gfx1250__) && __has_builtin(__builtin_amdgcn_tensor_load_to_lds)
#define HAVE_TDM 1
#else
#define HAVE_TDM 0
#endif

__device__ __forceinline__ unsigned short f2bf(float f) {
  unsigned int u = __float_as_uint(f);
  u += 0x7FFFu + ((u >> 16) & 1u);        // round-to-nearest-even
  return (unsigned short)(u >> 16);
}

#if HAVE_TDM
// Issue a TDM load of a 2D bf16 tile [tile_d1 rows][tile_d0 elems] from a
// row-major tensor (row stride d0_stride elems) into LDS at lds_addr.
// TDM packs tile rows contiguously in LDS -> exactly [tile_d1][tile_d0].
__device__ __forceinline__ void tdm_load_2d_bf16(
    unsigned lds_addr, const void* gptr,
    unsigned tensor_d0, unsigned tensor_d1,
    unsigned tile_d0, unsigned tile_d1, unsigned d0_stride) {
  unsigned long long ga = (unsigned long long)gptr;
  u32x4 g0;
  g0[0] = 1u;                                          // count=1, user mode
  g0[1] = lds_addr;                                    // LDS byte address
  g0[2] = (unsigned)(ga & 0xffffffffu);                // global_addr[31:0]
  g0[3] = (unsigned)((ga >> 32) & 0x01ffffffu) | (2u << 30);  // [56:32] | type=2
  i32x8 g1;
  g1[0] = (int)(1u << 16);                             // data_size=1 (2 bytes)
  g1[1] = (int)((tensor_d0 & 0xffffu) << 16);          // tensor_dim0[15:0]
  g1[2] = (int)((tensor_d0 >> 16) | ((tensor_d1 & 0xffffu) << 16));
  g1[3] = (int)((tensor_d1 >> 16) | (tile_d0 << 16));  // tile_dim0
  g1[4] = (int)(tile_d1 & 0xffffu);                    // tile_dim1 (tile_dim2=0)
  g1[5] = (int)d0_stride;                              // tensor_dim0_stride[31:0]
  g1[6] = 0;                                           // stride[47:32], dim1_stride lo
  g1[7] = 0;
  i32x4 g2 = {0, 0, 0, 0};
  i32x4 g3 = {0, 0, 0, 0};
#if __clang_major__ >= 23
  i32x8 g4 = {0, 0, 0, 0, 0, 0, 0, 0};
  __builtin_amdgcn_tensor_load_to_lds(g0, g1, g2, g3, g4, 0);
#else
  __builtin_amdgcn_tensor_load_to_lds(g0, g1, g2, g3, 0);
#endif
}
#endif

// ---------------------------------------------------------------- cast f32 -> bf16 (4 elems/thread, exact grids)
__global__ __launch_bounds__(256) void cast_f32_to_bf16(const float* __restrict__ in,
                                                        unsigned short* __restrict__ out) {
  size_t i = (size_t)blockIdx.x * 256 + threadIdx.x;
  f4 v = ((const f4*)in)[i];
  us4 o;
  o.x = f2bf(v.x); o.y = f2bf(v.y); o.z = f2bf(v.z); o.w = f2bf(v.w);
  ((us4*)out)[i] = o;
}

// ---------------------------------------------------------------- zero f32 (float4/thread, exact grids)
__global__ __launch_bounds__(256) void zero_f32(float* __restrict__ p) {
  size_t i = (size_t)blockIdx.x * 256 + threadIdx.x;
  f4 z = {0.f, 0.f, 0.f, 0.f};
  ((f4*)p)[i] = z;
}

// ---------------------------------------------------------------- bf16 WMMA GEMM
// C[M x Nn] = A[M x K](bf16, row-major) * B[Nn x K]^T(bf16, row-major) + bias[Nn]
// block tile 128x128, 256 threads (8 waves), wave tile 32x64 = 2x4 wmma tiles.
// Tiles are staged global->LDS by the Tensor Data Mover (wave 0 issues the
// descriptors; TENSORcnt + workgroup barrier order the double buffer).
__global__ __launch_bounds__(256) void gemm_bf16_wmma(
    const unsigned short* __restrict__ A,
    const unsigned short* __restrict__ Bm,
    const float* __restrict__ bias,
    float* __restrict__ C,
    int M, int Nn, int K) {
  __shared__ __align__(64) unsigned short As[2][128][32];
  __shared__ __align__(64) unsigned short Bs[2][128][32];

  const int tid  = threadIdx.x;
  const int lane = tid & 31;
  const int w    = tid >> 5;
  const int wm   = w & 3;           // M quadrant (32 rows)
  const int wn   = w >> 2;          // N half   (64 cols)
  const int m0   = blockIdx.x * 128;
  const int n0   = blockIdx.y * 128;

  v8f acc[2][4] = {};
  const int KT = K >> 5;

#if HAVE_TDM
  // ---------------- TDM-staged double buffer ----------------
  if (w == 0) {
    tdm_load_2d_bf16((unsigned)(uintptr_t)&As[0][0][0],
                     A + (size_t)m0 * K, (unsigned)K, (unsigned)M,
                     32u, 128u, (unsigned)K);
    tdm_load_2d_bf16((unsigned)(uintptr_t)&Bs[0][0][0],
                     Bm + (size_t)n0 * K, (unsigned)K, (unsigned)Nn,
                     32u, 128u, (unsigned)K);
    __builtin_amdgcn_s_wait_tensorcnt(0);
  }
  __syncthreads();

  for (int kt = 0; kt < KT; ++kt) {
    const int cur = kt & 1;
    if (w == 0 && kt + 1 < KT) {
      const int nxt = cur ^ 1;
      const int kO  = (kt + 1) << 5;
      tdm_load_2d_bf16((unsigned)(uintptr_t)&As[nxt][0][0],
                       A + (size_t)m0 * K + kO, (unsigned)K, (unsigned)M,
                       32u, 128u, (unsigned)K);
      tdm_load_2d_bf16((unsigned)(uintptr_t)&Bs[nxt][0][0],
                       Bm + (size_t)n0 * K + kO, (unsigned)K, (unsigned)Nn,
                       32u, 128u, (unsigned)K);
    }
    // compute on current buffer
    {
      const int rl = lane & 15;
      const int kh = (lane >> 4) * 16;
      v16bf afr[2], bfr[4];
#pragma unroll
      for (int r = 0; r < 2; ++r)
        afr[r] = *(const v16bf*)&As[cur][wm * 32 + r * 16 + rl][kh];
#pragma unroll
      for (int c = 0; c < 4; ++c)
        bfr[c] = *(const v16bf*)&Bs[cur][wn * 64 + c * 16 + rl][kh];
#pragma unroll
      for (int r = 0; r < 2; ++r)
#pragma unroll
        for (int c = 0; c < 4; ++c)
          acc[r][c] = __builtin_amdgcn_wmma_f32_16x16x32_bf16(
              false, afr[r], false, bfr[c], (short)0, acc[r][c], false, false);
    }
    if (w == 0 && kt + 1 < KT) __builtin_amdgcn_s_wait_tensorcnt(0);
    __syncthreads();
  }
#else
  // ---------------- fallback: register-staged double buffer ----------------
  const int lr = tid >> 1;          // 0..127 row of the staged tile
  const int lk = (tid & 1) * 16;    // 0 or 16 within the 32-wide K slab
  us8 ar[2], br[2];
  {
    const unsigned short* ap = A  + (size_t)(m0 + lr) * K + lk;
    const unsigned short* bp = Bm + (size_t)(n0 + lr) * K + lk;
    ar[0] = *(const us8*)ap; ar[1] = *(const us8*)(ap + 8);
    br[0] = *(const us8*)bp; br[1] = *(const us8*)(bp + 8);
    *(us8*)&As[0][lr][lk] = ar[0]; *(us8*)&As[0][lr][lk + 8] = ar[1];
    *(us8*)&Bs[0][lr][lk] = br[0]; *(us8*)&Bs[0][lr][lk + 8] = br[1];
  }
  __syncthreads();
  for (int kt = 0; kt < KT; ++kt) {
    const int cur = kt & 1;
    if (kt + 1 < KT) {
      const int kO = (kt + 1) << 5;
      const unsigned short* ap = A  + (size_t)(m0 + lr) * K + kO + lk;
      const unsigned short* bp = Bm + (size_t)(n0 + lr) * K + kO + lk;
      __builtin_prefetch(ap + 32, 0, 1);
      ar[0] = *(const us8*)ap; ar[1] = *(const us8*)(ap + 8);
      br[0] = *(const us8*)bp; br[1] = *(const us8*)(bp + 8);
    }
    {
      const int rl = lane & 15;
      const int kh = (lane >> 4) * 16;
      v16bf afr[2], bfr[4];
#pragma unroll
      for (int r = 0; r < 2; ++r)
        afr[r] = *(const v16bf*)&As[cur][wm * 32 + r * 16 + rl][kh];
#pragma unroll
      for (int c = 0; c < 4; ++c)
        bfr[c] = *(const v16bf*)&Bs[cur][wn * 64 + c * 16 + rl][kh];
#pragma unroll
      for (int r = 0; r < 2; ++r)
#pragma unroll
        for (int c = 0; c < 4; ++c)
          acc[r][c] = __builtin_amdgcn_wmma_f32_16x16x32_bf16(
              false, afr[r], false, bfr[c], (short)0, acc[r][c], false, false);
    }
    __syncthreads();
    if (kt + 1 < KT) {
      const int nxt = cur ^ 1;
      *(us8*)&As[nxt][lr][lk] = ar[0]; *(us8*)&As[nxt][lr][lk + 8] = ar[1];
      *(us8*)&Bs[nxt][lr][lk] = br[0]; *(us8*)&Bs[nxt][lr][lk + 8] = br[1];
      __syncthreads();
    }
  }
#endif

  // epilogue: C/D layout -> VGPR i: M = i + 8*(lane>=16), N = lane&15
  const int rl = lane & 15;
  const int rh = (lane >> 4) * 8;
#pragma unroll
  for (int r = 0; r < 2; ++r) {
    const int gm = m0 + wm * 32 + r * 16 + rh;
#pragma unroll
    for (int c = 0; c < 4; ++c) {
      const int gn = n0 + wn * 64 + c * 16 + rl;
      const float bv = bias[gn];
      float* cp = C + (size_t)gm * Nn + gn;
#pragma unroll
      for (int i = 0; i < 8; ++i)
        cp[(size_t)i * Nn] = acc[r][c][i] + bv;
    }
  }
}

// ---------------------------------------------------------------- per-head decaying-cosine kernels
// Rg[h][j] = ker_norm[h][G-1-j] for j<G, 0 for j>=G  (reversed + zero padded)
__global__ __launch_bounds__(256) void build_kernels(
    const float* __restrict__ freq, const float* __restrict__ damp,
    const float* __restrict__ phase, unsigned short* __restrict__ Rg) {
  const int h = blockIdx.x;
  const int t = threadIdx.x;
  const float alpha = log1pf(expf(damp[h]));   // softplus
  const float om = freq[h], ph = phase[h];
  __shared__ float red[256];
  float p = 0.f;
  for (int i = t; i < Gc; i += 256) {
    float v = expf(-alpha * (float)i) * cosf(om * (float)i + ph);
    p += fabsf(v);
  }
  red[t] = p;
  __syncthreads();
  for (int s = 128; s > 0; s >>= 1) {
    if (t < s) red[t] += red[t + s];
    __syncthreads();
  }
  const float inv = 1.0f / fmaxf(red[0], 1e-8f);
  for (int i = t; i < Gc; i += 256) {
    float v = expf(-alpha * (float)i) * cosf(om * (float)i + ph) * inv;
    Rg[h * 2 * Gc + (Gc - 1 - i)] = f2bf(v);
  }
  for (int i = Gc + t; i < 2 * Gc; i += 256) Rg[h * 2 * Gc + i] = 0;
}

// ---------------------------------------------------------------- deposit = (elu(k)+1)*v, scatter-add into field
__global__ __launch_bounds__(256) void deposit_field(
    const float* __restrict__ qkvg, float* __restrict__ field) {
  size_t idx = (size_t)blockIdx.x * 256 + threadIdx.x;   // over B*N*D
  int c  = (int)(idx % Dc);
  size_t bn = idx / Dc;
  int n = (int)(bn % Nc);
  int b = (int)(bn / Nc);
  const float* row = qkvg + bn * (size_t)(4 * Dc);
  float kx = row[Dc + c], vx = row[2 * Dc + c];
  float kf = kx > 0.f ? kx + 1.f : __expf(kx);
  float dep = kf * vx;
  float fp = fminf(fmaxf((float)n * STRIDEF, 0.f), (float)(Gc - 2));
  int lo = (int)fp;
  lo = lo < 0 ? 0 : (lo > Gc - 2 ? Gc - 2 : lo);
  float frac = fminf(fmaxf(fp - (float)lo, 0.f), 1.f);
  int h = c >> 6, d = c & 63;
  float* fb = field + (((size_t)b * Hc + h) * Gc) * HDc + d;
  atomicAdd(fb + (size_t)lo * HDc, dep * (1.f - frac));
  atomicAdd(fb + (size_t)(lo + 1) * HDc, dep * frac);
}

// ---------------------------------------------------------------- causal conv via Toeplitz WMMA
// conv[b,h,g,d] = sum_{t<=g} ker[h][g-t] * field[b,h,t,d]
// block = one (b,h) pair x 128 g-rows; wave w owns rows [g0, g0+16) x HD cols
__global__ __launch_bounds__(256) void conv_wmma(
    const float* __restrict__ field,        // [B*H][G][HD]
    const unsigned short* __restrict__ Rg,  // [H][2G] reversed bf16 kernel
    float* __restrict__ conv) {             // [B*H][G][HD]
  __shared__ __align__(64) unsigned short Rl[2 * Gc];       // 8 KB
  __shared__ __align__(64) unsigned short Fs[2][HDc][32];   // 8 KB, transposed bf16 field tile

  const int tid  = threadIdx.x;
  const int lane = tid & 31;
  const int w    = tid >> 5;
  const int pair = blockIdx.x >> 4;            // b*H + h
  const int h    = pair & (Hc - 1);
  const int blk  = blockIdx.x & 15;
  const int g0   = blk * 128 + w * 16;
  const float* fbase = field + (size_t)pair * Gc * HDc;
  float*       obase = conv  + (size_t)pair * Gc * HDc;

  for (int i = tid; i < 2 * Gc; i += 256) Rl[i] = Rg[h * 2 * Gc + i];

  const int NT = blk * 4 + 4;                  // t0 = 0 .. blk*128+96 step 32
  const int tl = tid >> 3;                     // 0..31 : t within tile
  const int d0 = (tid & 7) * 8;                // 0..56 : d chunk
  float fr[8];
  {
    const float* fp = fbase + (size_t)tl * HDc + d0;
#pragma unroll
    for (int j = 0; j < 8; ++j) fr[j] = fp[j];
#pragma unroll
    for (int j = 0; j < 8; ++j) Fs[0][d0 + j][tl] = f2bf(fr[j]);
  }
  __syncthreads();

  v8f acc[4] = {};
  for (int s = 0; s < NT; ++s) {
    const int cur = s & 1;
    const int t0  = s * 32;
    if (s + 1 < NT) {
      const float* fp = fbase + (size_t)((s + 1) * 32 + tl) * HDc + d0;
#pragma unroll
      for (int j = 0; j < 8; ++j) fr[j] = fp[j];
    }
    if (t0 <= g0 + 15) {                       // triangular: skip tiles above diagonal
      const int rl = lane & 15;
      const int kh = lane >> 4;
      const int j0 = (Gc - 1) - (g0 + rl) + t0 + kh * 16;  // ascending contiguous R read
      union { v16bf v; unsigned short u[16]; } au;
#pragma unroll
      for (int jj = 0; jj < 16; ++jj) au.u[jj] = Rl[j0 + jj];
#pragma unroll
      for (int c = 0; c < 4; ++c) {
        v16bf bfr = *(const v16bf*)&Fs[cur][c * 16 + rl][kh * 16];
        acc[c] = __builtin_amdgcn_wmma_f32_16x16x32_bf16(
            false, au.v, false, bfr, (short)0, acc[c], false, false);
      }
    }
    __syncthreads();
    if (s + 1 < NT) {
#pragma unroll
      for (int j = 0; j < 8; ++j) Fs[cur ^ 1][d0 + j][tl] = f2bf(fr[j]);
      __syncthreads();
    }
  }

  const int rl = lane & 15;
  const int rh = (lane >> 4) * 8;
#pragma unroll
  for (int c = 0; c < 4; ++c) {
    const int d = c * 16 + rl;
#pragma unroll
    for (int i = 0; i < 8; ++i)
      obase[(size_t)(g0 + rh + i) * HDc + d] = acc[c][i];
  }
}

// ---------------------------------------------------------------- softmax of 16x16 coupling
__global__ __launch_bounds__(256) void coup_softmax(const float* __restrict__ fc,
                                                    float* __restrict__ out) {
  __shared__ float m[16][16];
  int t = threadIdx.x;
  m[t >> 4][t & 15] = fc[t];
  __syncthreads();
  int r = t >> 4, c = t & 15;
  float mx = m[r][0];
#pragma unroll
  for (int j = 1; j < 16; ++j) mx = fmaxf(mx, m[r][j]);
  float s = 0.f;
#pragma unroll
  for (int j = 0; j < 16; ++j) s += __expf(m[r][j] - mx);
  out[t] = __expf(m[r][c] - mx) / s;
}

// ---------------------------------------------------------------- head coupling mix: out[b,i,g,d] = sum_j coup[i,j]*conv[b,j,g,d]
__global__ __launch_bounds__(256) void mix_heads(const float* __restrict__ conv,
                                                 const float* __restrict__ coup,
                                                 float* __restrict__ out) {
  __shared__ float cs[16][16];
  int t = threadIdx.x;
  cs[t >> 4][t & 15] = coup[t];
  __syncthreads();
  size_t idx = (size_t)blockIdx.x * 256 + t;       // over B*G*HD
  size_t gd  = idx % ((size_t)Gc * HDc);
  size_t b   = idx / ((size_t)Gc * HDc);
  const size_t hstride = (size_t)Gc * HDc;
  const float* cp = conv + b * Hc * hstride + gd;
  float vj[16];
#pragma unroll
  for (int j = 0; j < 16; ++j) vj[j] = cp[(size_t)j * hstride];
  float* op = out + b * Hc * hstride + gd;
#pragma unroll
  for (int i = 0; i < 16; ++i) {
    float s = 0.f;
#pragma unroll
    for (int j = 0; j < 16; ++j) s += cs[i][j] * vj[j];
    op[(size_t)i * hstride] = s;
  }
}

// ---------------------------------------------------------------- gather + q_feat * gate -> pre (bf16, [B*N][D])
__global__ __launch_bounds__(256) void gather_gate(
    const float* __restrict__ qkvg, const float* __restrict__ fieldc,
    unsigned short* __restrict__ pre) {
  size_t idx = (size_t)blockIdx.x * 256 + threadIdx.x;  // over B*N*D
  int c  = (int)(idx % Dc);
  size_t bn = idx / Dc;
  int n = (int)(bn % Nc);
  int b = (int)(bn / Nc);
  const float* row = qkvg + bn * (size_t)(4 * Dc);
  float qv = row[c], gr = row[3 * Dc + c];
  float qf = qv > 0.f ? qv + 1.f : __expf(qv);          // elu(q)+1
  float gate = 1.f / (1.f + __expf(-gr));               // sigmoid
  float fp = fminf(fmaxf((float)n * STRIDEF, 0.f), (float)(Gc - 2));
  int lo = (int)fp;
  lo = lo < 0 ? 0 : (lo > Gc - 2 ? Gc - 2 : lo);
  float frac = fminf(fmaxf(fp - (float)lo, 0.f), 1.f);
  int h = c >> 6, d = c & 63;
  const float* fb = fieldc + (((size_t)b * Hc + h) * Gc) * HDc + d;
  float glo = fb[(size_t)lo * HDc];
  float ghi = fb[(size_t)(lo + 1) * HDc];
  float val = qf * gate * ((1.f - frac) * glo + frac * ghi);
  pre[idx] = f2bf(val);
}

// ---------------------------------------------------------------- host launcher
extern "C" void kernel_launch(void* const* d_in, const int* in_sizes, int n_in,
                              void* d_out, int out_size, void* d_ws, size_t ws_size,
                              hipStream_t stream) {
  (void)in_sizes; (void)n_in; (void)out_size; (void)ws_size;
  const float* x      = (const float*)d_in[0];
  const float* Wqkvg  = (const float*)d_in[1];
  const float* bqkvg  = (const float*)d_in[2];
  const float* freq   = (const float*)d_in[3];
  const float* damp   = (const float*)d_in[4];
  const float* phase  = (const float*)d_in[5];
  const float* fcoup  = (const float*)d_in[6];
  const float* Wout   = (const float*)d_in[7];
  const float* bout   = (const float*)d_in[8];
  float* out = (float*)d_out;

  char* ws = (char*)d_ws;
  size_t off = 0;
  auto alloc = [&](size_t bytes) -> char* {
    char* p = ws + off;
    off += (bytes + 255) & ~(size_t)255;
    return p;
  };
  unsigned short* xbf   = (unsigned short*)alloc((size_t)BN * Dc * 2);       // 16 MB
  unsigned short* w1bf  = (unsigned short*)alloc((size_t)4 * Dc * Dc * 2);   // 8 MB
  unsigned short* wobf  = (unsigned short*)alloc((size_t)Dc * Dc * 2);       // 2 MB
  float*          qkvg  = (float*)alloc((size_t)BN * 4 * Dc * 4);            // 128 MB
  float*          field = (float*)alloc((size_t)Bc * Hc * Gc * HDc * 4);     // 32 MB (reused as field_c)
  float*          conv  = (float*)alloc((size_t)Bc * Hc * Gc * HDc * 4);     // 32 MB
  unsigned short* Rg    = (unsigned short*)alloc((size_t)Hc * 2 * Gc * 2);   // 128 KB
  float*          coup  = (float*)alloc((size_t)Hc * Hc * 4);                // tiny
  unsigned short* prebf = (unsigned short*)alloc((size_t)BN * Dc * 2);       // 16 MB

  // 1) casts
  cast_f32_to_bf16<<<(size_t)BN * Dc / 1024, 256, 0, stream>>>(x, xbf);
  cast_f32_to_bf16<<<(size_t)4 * Dc * Dc / 1024, 256, 0, stream>>>(Wqkvg, w1bf);
  cast_f32_to_bf16<<<(size_t)Dc * Dc / 1024, 256, 0, stream>>>(Wout, wobf);

  // 2) zero the field accumulator
  zero_f32<<<(size_t)Bc * Hc * Gc * HDc / 1024, 256, 0, stream>>>(field);

  // 3) QKVG GEMM: (8192x1024)x(4096x1024)^T + bias (TDM-staged WMMA)
  gemm_bf16_wmma<<<dim3(BN / 128, (4 * Dc) / 128), 256, 0, stream>>>(
      xbf, w1bf, bqkvg, qkvg, BN, 4 * Dc, Dc);

  // 4) wave kernels (reversed, L1-normalized, bf16) + coupling softmax
  build_kernels<<<Hc, 256, 0, stream>>>(freq, damp, phase, Rg);
  coup_softmax<<<1, 256, 0, stream>>>(fcoup, coup);

  // 5) deposit scatter into field
  deposit_field<<<(size_t)BN * Dc / 256, 256, 0, stream>>>(qkvg, field);

  // 6) causal convolution (Toeplitz WMMA)
  conv_wmma<<<Bc * Hc * 16, 256, 0, stream>>>(field, Rg, conv);

  // 7) head coupling mix (writes into field buffer, now dead)
  mix_heads<<<(size_t)Bc * Gc * HDc / 256, 256, 0, stream>>>(conv, coup, field);

  // 8) gather + q_feat * gate -> bf16 pre-activation
  gather_gate<<<(size_t)BN * Dc / 256, 256, 0, stream>>>(qkvg, field, prebf);

  // 9) output GEMM: (8192x1024)x(1024x1024)^T + bout
  gemm_bf16_wmma<<<dim3(BN / 128, Dc / 128), 256, 0, stream>>>(
      prebf, wobf, bout, out, BN, Dc, Dc);
}
